// STx_encoder_59407987638477
// MI455X (gfx1250) — compile-verified, matrix operand order checked by reading, not run on hardware
//
#include <hip/hip_runtime.h>

typedef __bf16 bf16;
typedef __attribute__((ext_vector_type(16))) __bf16 v16bf;
typedef __attribute__((ext_vector_type(8)))  __bf16 v8bf;
typedef __attribute__((ext_vector_type(8)))  float  v8f;
typedef __attribute__((ext_vector_type(4)))  float  v4f;
typedef __attribute__((ext_vector_type(2)))  float  v2f;

#define N_NODES 10000
#define CIN     256
#define OUTC    256
#define KNBR    16
#define NTILES  625   // N_NODES / 16

// ---------------------------------------------------------------------------
// prep: convert x -> bf16, compute squared row norms
// ---------------------------------------------------------------------------
__global__ void prep_x_kernel(const float* __restrict__ x,
                              bf16* __restrict__ xb,
                              float* __restrict__ n2) {
  __shared__ float red[256];
  int i = blockIdx.x, t = threadIdx.x;
  float v = x[(size_t)i * CIN + t];
  xb[(size_t)i * CIN + t] = (bf16)v;
  red[t] = v * v;
  __syncthreads();
  for (int s = 128; s > 0; s >>= 1) {
    if (t < s) red[t] += red[t + s];
    __syncthreads();
  }
  if (t == 0) n2[i] = red[0];
}

// ---------------------------------------------------------------------------
// pack weights into WMMA B-fragment layout:
// fragment f = nt*KT + kt ; per lane 16 contiguous halfs:
//   K = kt*32 + j + 8*(j>=8) + 8*(lane>=16),  N = nt*16 + (lane&15)
// ---------------------------------------------------------------------------
__global__ void pack_w_kernel(const float* __restrict__ Wq,
                              const float* __restrict__ Wk,
                              const float* __restrict__ Wl1,
                              const float* __restrict__ Wl2,
                              bf16* __restrict__ Wq_p, bf16* __restrict__ Wk_p,
                              bf16* __restrict__ Wl1_p, bf16* __restrict__ Wl2_p) {
  int b = blockIdx.x, lane = threadIdx.x;
  const float* W; bf16* D; int KT, f;
  if (b < 256)      { W = Wq;  D = Wq_p;  KT = 16; f = b; }
  else if (b < 512) { W = Wk;  D = Wk_p;  KT = 16; f = b - 256; }
  else if (b < 768) { W = Wl1; D = Wl1_p; KT = 16; f = b - 512; }
  else              { W = Wl2; D = Wl2_p; KT = 8;  f = b - 768; }
  int kt  = f % KT;
  int nt  = f / KT;
  int col = nt * 16 + (lane & 15);
  int hi8 = (lane >= 16) ? 8 : 0;
  bf16* dst = D + ((size_t)f * 32 + lane) * 16;
  #pragma unroll
  for (int j = 0; j < 16; ++j) {
    int K = kt * 32 + j + ((j >= 8) ? 8 : 0) + hi8;
    dst[j] = (bf16)W[(size_t)K * OUTC + col];
  }
}

// ---------------------------------------------------------------------------
// kNN: one block per 16-query tile. Query A-fragments resident in VGPRs.
// Candidate block (128 rows) cooperatively staged into LDS with coalesced
// b128 loads; WMMA loop reads B fragments from LDS with a 2-chain
// accumulator + 1-deep prefetch pinned by sched_barrier(0).
// ---------------------------------------------------------------------------
#define XC_S    260                       // f32 row stride (pad: conflict-free)
#define KSM_XC  0                         // [128][260] f32   = 133120
#define KSM_DT  133120                    // [8][16][17] f32  = 8704
#define KSM_LD  141824                    // [8][16][16] f32  = 8192
#define KSM_LI  150016                    // [8][16][16] int  = 8192
#define KSM_TOT 158208

__global__ void __launch_bounds__(256)
knn_kernel(const float* __restrict__ x, const float* __restrict__ n2,
           int* __restrict__ nbr) {
  extern __shared__ __align__(16) char ksm[];
  float* Xc = (float*)(ksm + KSM_XC);
  float* dt = (float*)(ksm + KSM_DT);
  float* ld = (float*)(ksm + KSM_LD);
  int*   li = (int*)(ksm + KSM_LI);

  const int tid  = threadIdx.x;
  const int lane = tid & 31;
  const int w    = tid >> 5;
  const int col  = lane & 15;
  const int hi   = lane >> 4;
  const int q0   = blockIdx.x * 16;

  // Resident A fragments: VGPR0 K=k0+2*hi, VGPR1 K=k0+2*hi+1, M=lane&15
  v2f afrag[64];
  {
    const float* qrow = x + (size_t)(q0 + col) * CIN + 2 * hi;
    #pragma unroll
    for (int kt = 0; kt < 64; ++kt)
      afrag[kt] = *(const v2f*)(qrow + 4 * kt);
  }
  float n2q[8];
  #pragma unroll
  for (int r = 0; r < 8; ++r) n2q[r] = n2[q0 + r + 8 * hi];

  if (lane < 16) {
    #pragma unroll
    for (int s = 0; s < 16; ++s) {
      ld[(w * 16 + lane) * 16 + s] = __builtin_inff();
      li[(w * 16 + lane) * 16 + s] = 0;
    }
  }

  for (int base = 0; base < NTILES; base += 8) {
    int ntile = NTILES - base; if (ntile > 8) ntile = 8;
    int nrows = ntile * 16;
    __syncthreads();
    // stage candidate rows [base*16, base*16+nrows) : coalesced b128
    for (int it = 0; it < 32; ++it) {
      int flat = tid + 256 * it;          // b128 units, 64 per row
      int row  = flat >> 6;
      int cq   = flat & 63;
      if (row < nrows)
        *(v4f*)(Xc + row * XC_S + cq * 4) =
            *(const v4f*)(x + (size_t)(base * 16 + row) * CIN + cq * 4);
    }
    __syncthreads();

    if (w < ntile) {
      const int ct   = base + w;
      const int crow = ct * 16 + col;
      const float* Bl = Xc + (w * 16 + col) * XC_S + 2 * hi;
      float n2c = n2[crow];

      v8f acc0 = {}, acc1 = {};
      v2f b0 = *(const v2f*)(Bl + 0);
      v2f b1 = *(const v2f*)(Bl + 128);
      #pragma unroll
      for (int kt = 0; kt < 32; ++kt) {
        v2f b0n = b0, b1n = b1;
        if (kt + 1 < 32) {
          b0n = *(const v2f*)(Bl + 4 * (kt + 1));
          b1n = *(const v2f*)(Bl + 4 * (kt + 1) + 128);
        }
        __builtin_amdgcn_sched_barrier(0);  // keep prefetch above the WMMAs
        acc0 = __builtin_amdgcn_wmma_f32_16x16x4_f32(
            false, afrag[kt], false, b0, (short)0, acc0, false, false);
        acc1 = __builtin_amdgcn_wmma_f32_16x16x4_f32(
            false, afrag[32 + kt], false, b1, (short)0, acc1, false, false);
        b0 = b0n; b1 = b1n;
      }

      #pragma unroll
      for (int r = 0; r < 8; ++r) {
        int m = r + 8 * hi;
        float d = n2q[r] + n2c - 2.0f * (acc0[r] + acc1[r]);
        if (crow == q0 + m) d = __builtin_inff();   // no self loops
        dt[(w * 16 + m) * 17 + col] = d;
      }
      if (lane < 16) {
        float* Lrow = ld + (w * 16 + lane) * 16;
        int*   Irow = li + (w * 16 + lane) * 16;
        for (int c = 0; c < 16; ++c) {
          float d = dt[(w * 16 + lane) * 17 + c];
          if (d < Lrow[15]) {
            int cand = ct * 16 + c;
            int p = 15;
            while (p > 0 && Lrow[p - 1] > d) {
              Lrow[p] = Lrow[p - 1];
              Irow[p] = Irow[p - 1];
              --p;
            }
            Lrow[p] = d;
            Irow[p] = cand;
          }
        }
      }
    }
  }
  __syncthreads();
  if (tid < 16) {
    int row = tid;
    int p[8] = {0, 0, 0, 0, 0, 0, 0, 0};
    for (int s = 0; s < 16; ++s) {
      float best = __builtin_inff(); int bw = 0;
      #pragma unroll
      for (int ww = 0; ww < 8; ++ww) {
        float dv = ld[(ww * 16 + row) * 16 + p[ww]];
        if (dv < best) { best = dv; bw = ww; }
      }
      nbr[(size_t)(q0 + row) * KNBR + s] = li[(bw * 16 + row) * 16 + p[bw]];
      ++p[bw];
    }
  }
}

// ---------------------------------------------------------------------------
// bf16 WMMA 16x16 tile GEMM over LDS A (row stride sA halfs, column XOR cxor)
// times packed weights Wp (KT k-tiles of 32). 2-deep software pipeline with
// sched_barrier(0) pinning the prefetch above the current WMMA.
// ---------------------------------------------------------------------------
template <int KT>
__device__ __forceinline__ v8f mm_tile(const bf16* Abase, int sA, int cxor,
                                       const bf16* __restrict__ Wp,
                                       int mt, int nt, int lane) {
  const int col = lane & 15;
  const int hi8 = (lane >= 16) ? 8 : 0;
  const bf16* arow = Abase + (size_t)(mt * 16 + col) * sA;
  const bf16* wf   = Wp + ((size_t)nt * KT * 32 + lane) * 16;

  v16bf a[2], b[2];
  {
    v8bf lo = *(const v8bf*)(arow + ((0 + hi8) ^ cxor));
    v8bf hp = *(const v8bf*)(arow + ((16 + hi8) ^ cxor));
    #pragma unroll
    for (int j = 0; j < 8; ++j) { a[0][j] = lo[j]; a[0][j + 8] = hp[j]; }
    b[0] = *(const v16bf*)(wf);
  }
  v8f acc = {};
  #pragma unroll
  for (int kt = 0; kt < KT; ++kt) {
    const int cur = kt & 1, nxt = cur ^ 1;
    if (kt + 1 < KT) {
      int k0 = (kt + 1) * 32;
      v8bf lo = *(const v8bf*)(arow + ((k0 + hi8) ^ cxor));
      v8bf hp = *(const v8bf*)(arow + ((k0 + 16 + hi8) ^ cxor));
      #pragma unroll
      for (int j = 0; j < 8; ++j) { a[nxt][j] = lo[j]; a[nxt][j + 8] = hp[j]; }
      b[nxt] = *(const v16bf*)(wf + (size_t)(kt + 1) * 32 * 16);
    }
    __builtin_amdgcn_sched_barrier(0);  // prefetch stays above current WMMA
    acc = __builtin_amdgcn_wmma_f32_16x16x32_bf16(
        false, a[cur], false, b[cur], (short)0, acc, false, false);
  }
  return acc;
}

// ---------------------------------------------------------------------------
// edge kernel: 1 block = 2 nodes = 32 edges (two 16-row M-tiles).
// Af = [x_j | x_i] (520-half padded rows -> conflict-free A fragments).
// Q/K -> ew (2x2 heads) -> H1 (relu + x_i, reads Af with col^256) -> H2,
// ew mixing via shfl_xor(1), per-node mean via shfl_xor(16).
// ---------------------------------------------------------------------------
#define AF_S   520
#define QB_S   264
#define SM_AF  0
#define SM_KB  (32 * AF_S * 2)                 // 33280
#define SM_QB  (SM_KB + 32 * QB_S * 2)         // 50176
#define SM_EW  (SM_QB + 32 * QB_S * 2)         // 67072
#define SM_JN  (SM_EW + 32 * 4 * 4)            // 67584
#define SM_TOT (SM_JN + 32 * 4)                // 67712

__global__ void __launch_bounds__(256)
edge_kernel(const bf16* __restrict__ xb, const int* __restrict__ nbr,
            const bf16* __restrict__ Wq_p, const bf16* __restrict__ Wk_p,
            const bf16* __restrict__ Wl1_p, const bf16* __restrict__ Wl2_p,
            const float* __restrict__ bq, const float* __restrict__ bk,
            const float* __restrict__ b1, const float* __restrict__ b2,
            float* __restrict__ out) {
  extern __shared__ __align__(16) char smem[];
  bf16*  Af   = (bf16*)(smem + SM_AF);    // [32][520]
  bf16*  Kb   = (bf16*)(smem + SM_KB);    // [32][264]
  bf16*  Qb   = (bf16*)(smem + SM_QB);    // [32][264], later reused as H1
  bf16*  H1b  = Qb;
  float* ew_s = (float*)(smem + SM_EW);   // [32][4]
  int*   jn   = (int*)(smem + SM_JN);     // [32]

  const int tid  = threadIdx.x;
  const int lane = tid & 31;
  const int w    = tid >> 5;
  const int col  = lane & 15;
  const int hi8  = (lane >= 16) ? 8 : 0;
  const int i0   = blockIdx.x * 2;

  if (tid < 32) jn[tid] = nbr[(size_t)(i0 + (tid >> 4)) * KNBR + (tid & 15)];
  __syncthreads();

  // Phase A: gather Af[e] = [ xb[j] | xb[i] ], u32 copies
  {
    const unsigned* xb32 = (const unsigned*)xb;
    #pragma unroll 4
    for (int it = 0; it < 32; ++it) {
      int flat = tid + 256 * it;          // 8192 dwords total
      int e  = flat >> 8;
      int cu = flat & 255;
      int node = (cu < 128) ? jn[e] : (i0 + (e >> 4));
      int srcc = (cu < 128) ? cu : (cu - 128);
      unsigned v = xb32[(size_t)node * 128 + srcc];
      *(unsigned*)(Af + (size_t)e * AF_S + cu * 2) = v;
    }
  }
  __syncthreads();

  // Phase B: Q and K (64 tile jobs over 8 waves)
  for (int t = 0; t < 8; ++t) {
    int job  = w + 8 * t;
    int tile = job & 31, mt = tile >> 4, nt = tile & 15;
    bool isQ = job < 32;
    v8f acc = mm_tile<16>(Af, AF_S, 0, isQ ? Wq_p : Wk_p, mt, nt, lane);
    float bias = isQ ? bq[nt * 16 + col] : bk[nt * 16 + col];
    bf16* Dst = isQ ? Qb : Kb;
    #pragma unroll
    for (int r = 0; r < 8; ++r)
      Dst[(size_t)(mt * 16 + r + hi8) * QB_S + nt * 16 + col] =
          (bf16)(acc[r] + bias);
  }
  __syncthreads();

  // Phase C: ew[e][h][g] = scale * <Q[e,h,:], K[e,g,:]>  (vector LDS reads)
  if (tid < 128) {
    int e = tid >> 2, h = (tid >> 1) & 1, g = tid & 1;
    const v8bf* qr = (const v8bf*)(Qb + (size_t)e * QB_S + h * 128);
    const v8bf* kr = (const v8bf*)(Kb + (size_t)e * QB_S + g * 128);
    float s = 0.f;
    #pragma unroll
    for (int d = 0; d < 16; ++d) {
      v8bf qv = qr[d], kv = kr[d];
      #pragma unroll
      for (int j = 0; j < 8; ++j) s += (float)qv[j] * (float)kv[j];
    }
    ew_s[e * 4 + h * 2 + g] = s * 0.0625f;   // 1/sqrt(256)
  }
  __syncthreads();

  // Phase D: H1 = relu([x_i|x_j] @ Wl1 + b1) + x_i  (col^256 view of Af)
  for (int t = 0; t < 4; ++t) {
    int tile = w + 8 * t;
    int mt = tile >> 4, nt = tile & 15;
    v8f acc = mm_tile<16>(Af, AF_S, 256, Wl1_p, mt, nt, lane);
    float bias = b1[nt * 16 + col];
    #pragma unroll
    for (int r = 0; r < 8; ++r) {
      int e = mt * 16 + r + hi8;
      float v = acc[r] + bias;
      v = v > 0.f ? v : 0.f;
      v += (float)Af[(size_t)e * AF_S + 256 + nt * 16 + col];   // + x_i
      H1b[(size_t)e * QB_S + nt * 16 + col] = (bf16)v;
    }
  }
  __syncthreads();

  // Phase E: H2 = H1 @ Wl2 + b2, ew mixing, per-node mean over 16 edges
  for (int t = 0; t < 4; ++t) {
    int tile = w + 8 * t;
    int mt = tile >> 4, nt = tile & 15;
    v8f acc = mm_tile<8>(H1b, QB_S, 0, Wl2_p, mt, nt, lane);
    float bias = b2[nt * 16 + col];
    int cglob = nt * 16 + col;
    float ssum = 0.f;
    #pragma unroll
    for (int r = 0; r < 8; ++r) {
      int e = mt * 16 + r + hi8;
      float hval  = acc[r] + bias;
      float other = __shfl_xor(hval, 1, 32);   // column c^1
      float msg;
      if ((cglob & 1) == 0)
        msg = hval * ew_s[e * 4 + 0] + other * ew_s[e * 4 + 2];
      else
        msg = other * ew_s[e * 4 + 1] + hval * ew_s[e * 4 + 3];
      ssum += msg;
    }
    float tot = ssum + __shfl_xor(ssum, 16, 32);  // fold both lane halves
    if (lane < 16)
      out[(size_t)(i0 + mt) * OUTC + cglob] = tot * (1.0f / 16.0f);
  }
}

// ---------------------------------------------------------------------------
extern "C" void kernel_launch(void* const* d_in, const int* in_sizes, int n_in,
                              void* d_out, int out_size, void* d_ws, size_t ws_size,
                              hipStream_t stream) {
  (void)in_sizes; (void)n_in; (void)out_size; (void)ws_size;
  const float* x   = (const float*)d_in[0];
  const float* Wl1 = (const float*)d_in[1];
  const float* b1  = (const float*)d_in[2];
  const float* Wl2 = (const float*)d_in[3];
  const float* b2  = (const float*)d_in[4];
  const float* Wq  = (const float*)d_in[5];
  const float* bq  = (const float*)d_in[6];
  const float* Wk  = (const float*)d_in[7];
  const float* bk  = (const float*)d_in[8];

  char* ws = (char*)d_ws;
  bf16*  xb    = (bf16*)ws;   ws += (size_t)N_NODES * CIN * 2;   // 5,120,000
  float* n2    = (float*)ws;  ws += 40192;                       // N f32, padded
  int*   nbr   = (int*)ws;    ws += (size_t)N_NODES * KNBR * 4;  // 640,000
  bf16*  Wq_p  = (bf16*)ws;   ws += 512 * 256 * 2;
  bf16*  Wk_p  = (bf16*)ws;   ws += 512 * 256 * 2;
  bf16*  Wl1_p = (bf16*)ws;   ws += 512 * 256 * 2;
  bf16*  Wl2_p = (bf16*)ws;   ws += 256 * 256 * 2;

  prep_x_kernel<<<N_NODES, 256, 0, stream>>>(x, xb, n2);
  pack_w_kernel<<<896, 32, 0, stream>>>(Wq, Wk, Wl1, Wl2,
                                        Wq_p, Wk_p, Wl1_p, Wl2_p);
  knn_kernel<<<NTILES, 256, KSM_TOT, stream>>>(x, n2, nbr);
  edge_kernel<<<N_NODES / 2, 256, SM_TOT, stream>>>(
      xb, nbr, Wq_p, Wk_p, Wl1_p, Wl2_p, bq, bk, b1, b2, (float*)d_out);
}
